// myGAT_1700807049281
// MI455X (gfx1250) — matrix operand order repaired and checked by reading, not compiled, not run on hardware
//
#include <hip/hip_runtime.h>
#include <hip/hip_bf16.h>
#include <cstddef>
#include <cstdint>

#define N_NODES 40000
#define N_EDGES 320000
#define N_ETYPES 5
#define EDGE_DIM 64
#define HID 64
#define HEADS 8
#define NCLS 16
#define HF (HID * HEADS) /* 512 */
#define NEGS 0.2f
#define ALPHA_R 0.05f

#define BM 64
#define BN 128
#define BK 32

typedef __attribute__((ext_vector_type(16))) __bf16 v16bf;
typedef __attribute__((ext_vector_type(8))) float v8f;
typedef __attribute__((ext_vector_type(8))) unsigned short us8;

union FragU { us8 u[2]; v16bf v; };

__device__ __forceinline__ unsigned short f2bf(float f) {
  unsigned u = __float_as_uint(f);
  u = u + 0x7FFFu + ((u >> 16) & 1u);      // round-to-nearest-even
  return (unsigned short)(u >> 16);
}

__device__ __forceinline__ unsigned fkey(float f) {  // monotonic float->uint
  unsigned u = __float_as_uint(f);
  return ((int)u >= 0) ? (u | 0x80000000u) : ~u;
}
__device__ __forceinline__ float dekey(unsigned k) {
  unsigned u = (k & 0x80000000u) ? (k & 0x7FFFFFFFu) : ~k;
  return __uint_as_float(u);
}
__device__ __forceinline__ float eluf(float x) { return x > 0.f ? x : (__expf(x) - 1.f); }

// ---------------- conversion kernels ----------------
__global__ void k_f32_to_bf16(const float* __restrict__ in, unsigned short* __restrict__ out, int n) {
  int i = blockIdx.x * blockDim.x + threadIdx.x;
  if (i < n) out[i] = f2bf(in[i]);
}

// W [K,N] f32 row-major  ->  Bt [N,K] bf16 row-major (i.e., B column-major)
__global__ void k_transpose_bf16(const float* __restrict__ W, unsigned short* __restrict__ Bt, int K, int N) {
  int i = blockIdx.x * blockDim.x + threadIdx.x;
  if (i >= N * K) return;
  int n = i / K, k = i - n * K;
  Bt[i] = f2bf(W[(size_t)k * N + n]);
}

__global__ void k_zero(float* __restrict__ p, int n) {
  int i = blockIdx.x * blockDim.x + threadIdx.x;
  if (i < n) p[i] = 0.f;
}

// ---------------- WMMA bf16 GEMM: C[M,N] = A[M,K] * B[K,N] (+bias) ----------------
__global__ __launch_bounds__(256) void k_gemm_bf16(
    const unsigned short* __restrict__ A,   // [M,K] bf16
    const unsigned short* __restrict__ Bt,  // [N,K] bf16 (B transposed)
    float* __restrict__ C,                  // [M,N] f32
    const float* __restrict__ bias,         // [N] or nullptr
    int M, int N, int K) {
  __shared__ unsigned short sA[BM * BK];    // 4 KB
  __shared__ unsigned short sB[BN * BK];    // 8 KB
  const int tid = threadIdx.x;
  const int lane = tid & 31;
  const int wave = tid >> 5;
  const int mTile = wave & 3;    // 4 M-tiles of 16
  const int nGroup = wave >> 2;  // 2 N-groups of 64
  const int mBase = blockIdx.x * BM;
  const int nBase = blockIdx.y * BN;
  const int half = lane >> 4;
  const int l16 = lane & 15;

  v8f acc[4] = {};
  us8 uz = {};

  for (int k0 = 0; k0 < K; k0 += BK) {
    { // stage A tile 64x32: 256 chunks of 8 bf16 (16B)
      int row = tid >> 2, ch = tid & 3;
      us8 val = uz;
      int gr = mBase + row;
      if (gr < M) val = *(const us8*)(A + (size_t)gr * K + k0 + ch * 8);
      *(us8*)(sA + row * BK + ch * 8) = val;
    }
    { // stage B tile 128x32: 512 chunks, 2 per thread
      int row = tid >> 1, chp = (tid & 1) * 2;
      us8 v0 = uz, v1 = uz;
      int gr = nBase + row;
      if (gr < N) {
        const unsigned short* bp = Bt + (size_t)gr * K + k0 + chp * 8;
        v0 = *(const us8*)bp;
        v1 = *(const us8*)(bp + 8);
      }
      *(us8*)(sB + row * BK + chp * 8) = v0;
      *(us8*)(sB + row * BK + chp * 8 + 8) = v1;
    }
    __syncthreads();
    // A frag per ISA 16-bit 16x32 layout: lanes0-15 row m K{0..7,16..23}, lanes16-31 K{8..15,24..31}
    FragU fa;
    {
      const unsigned short* ap = sA + (mTile * 16 + l16) * BK;
      fa.u[0] = *(const us8*)(ap + half * 8);
      fa.u[1] = *(const us8*)(ap + 16 + half * 8);
    }
#pragma unroll
    for (int t = 0; t < 4; ++t) {
      // B frag: lane -> col N=l16; lanes0-15 K{0..7,16..23}, lanes16-31 K{8..15,24..31}
      FragU fb;
      const unsigned short* bp = sB + (nGroup * 64 + t * 16 + l16) * BK + half * 8;
      fb.u[0] = *(const us8*)bp;
      fb.u[1] = *(const us8*)(bp + 16);
      acc[t] = __builtin_amdgcn_wmma_f32_16x16x32_bf16(
          false, fa.v, false, fb.v, (short)0, acc[t], false, false);
    }
    __syncthreads();
  }
#pragma unroll
  for (int t = 0; t < 4; ++t) {
    int col = nBase + nGroup * 64 + t * 16 + l16;
    if (col < N) {
      float badd = bias ? bias[col] : 0.f;
#pragma unroll
      for (int r = 0; r < 8; ++r) {   // C layout: VGPR r -> M = half*8 + r
        int m = mBase + mTile * 16 + half * 8 + r;
        if (m < M) C[(size_t)m * N + col] = acc[t][r] + badd;
      }
    }
  }
}

// ---------------- per-edge-type attention scalar table ----------------
// ee[t,h] = sum_d (eemb[t] @ We)[h*Ed+d] * ae[h,d]
__global__ void k_ee(const float* __restrict__ eemb, const float* __restrict__ We,
                     const float* __restrict__ ae, float* __restrict__ ee, int Ed, int H) {
  int t = blockIdx.x / H, h = blockIdx.x % H;
  int d = threadIdx.x;  // blockDim == Ed (64)
  __shared__ float red[EDGE_DIM];
  float ef = 0.f;
  for (int k = 0; k < Ed; ++k) ef += eemb[t * Ed + k] * We[(size_t)k * H * Ed + h * Ed + d];
  red[d] = ef * ae[h * Ed + d];
  __syncthreads();
  for (int s2 = Ed / 2; s2 > 0; s2 >>= 1) {
    if (d < s2) red[d] += red[d + s2];
    __syncthreads();
  }
  if (d == 0) ee[t * H + h] = red[0];
}

// el[n,h] = feat[n,h,:].al[h]; er likewise
__global__ void k_elr(const float* __restrict__ feat, const float* __restrict__ al,
                      const float* __restrict__ ar, float* __restrict__ el, float* __restrict__ er,
                      int Nn, int H, int F) {
  int i = blockIdx.x * blockDim.x + threadIdx.x;
  if (i >= Nn * H) return;
  int n = i / H, h = i - n * H;
  const float* fp = feat + (size_t)n * H * F + h * F;
  const float* alp = al + h * F;
  const float* arp = ar + h * F;
  float sl = 0.f, sr = 0.f;
  for (int f = 0; f < F; ++f) { float v = fp[f]; sl += v * alp[f]; sr += v * arp[f]; }
  el[i] = sl; er[i] = sr;
}

__global__ void k_init_ms(unsigned* __restrict__ mkey, float* __restrict__ ssum, int n) {
  int i = blockIdx.x * blockDim.x + threadIdx.x;
  if (i < n) { mkey[i] = 0u; ssum[i] = 0.f; }
}

// pass1: e-value + segment max (monotonic uint atomicMax)
__global__ void k_att1(const int* __restrict__ src, const int* __restrict__ dst,
                       const int* __restrict__ et, const float* __restrict__ el,
                       const float* __restrict__ er, const float* __restrict__ ee,
                       float* __restrict__ eval, unsigned* __restrict__ mkey, int E, int H) {
  int i = blockIdx.x * blockDim.x + threadIdx.x;
  if (i >= E * H) return;
  int e = i / H, h = i - e * H;
  float v = el[src[e] * H + h] + er[dst[e] * H + h] + ee[et[e] * H + h];
  v = v > 0.f ? v : NEGS * v;   // leaky relu
  eval[i] = v;
  atomicMax(&mkey[dst[e] * H + h], fkey(v));
}

// pass2: w = exp(e - m[dst]); segment sum
__global__ void k_att2(const int* __restrict__ dst, float* __restrict__ w,
                       const unsigned* __restrict__ mkey, float* __restrict__ ssum, int E, int H) {
  int i = blockIdx.x * blockDim.x + threadIdx.x;
  if (i >= E * H) return;
  int e = i / H, h = i - e * H;
  float m = dekey(mkey[dst[e] * H + h]);
  float ww = __expf(w[i] - m);
  w[i] = ww;
  atomicAdd(&ssum[dst[e] * H + h], ww);
}

// pass3: a = w/s (+attn residual blend), optionally save for next layer
__global__ void k_att3(const int* __restrict__ dst, float* __restrict__ a,
                       const float* __restrict__ ssum, const float* __restrict__ resIn,
                       float* __restrict__ resOut, int E, int H) {
  int i = blockIdx.x * blockDim.x + threadIdx.x;
  if (i >= E * H) return;
  int e = i / H, h = i - e * H;
  float v = a[i] / ssum[dst[e] * H + h];
  if (resIn) v = v * (1.f - ALPHA_R) + resIn[i] * ALPHA_R;
  a[i] = v;
  if (resOut) resOut[i] = v;
}

// aggregation for H*F = 512: one wave32 per edge, float4 gathers, L2 atomic scatter-add
__global__ __launch_bounds__(256) void k_agg512(const int* __restrict__ src,
                                                const int* __restrict__ dst,
                                                const float* __restrict__ feat,
                                                const float* __restrict__ a,
                                                float* __restrict__ rst, int E) {
  int eid = (blockIdx.x * blockDim.x + threadIdx.x) >> 5;
  int lane = threadIdx.x & 31;
  if (eid >= E) return;
  int sN = src[eid], dN = dst[eid];
  const float4* fp = (const float4*)(feat + (size_t)sN * HF);
  float* rp = rst + (size_t)dN * HF;
  const float* ap = a + (size_t)eid * HEADS;
  __builtin_prefetch(fp, 0, 3);
#pragma unroll
  for (int j = 0; j < 4; ++j) {
    int base = j * 128 + lane * 4;
    float av = ap[base >> 6];
    float4 v = fp[base >> 2];
    atomicAdd(rp + base + 0, av * v.x);
    atomicAdd(rp + base + 1, av * v.y);
    atomicAdd(rp + base + 2, av * v.z);
    atomicAdd(rp + base + 3, av * v.w);
  }
}

// final-layer aggregation (H=1, F=16)
__global__ void k_agg16(const int* __restrict__ src, const int* __restrict__ dst,
                        const float* __restrict__ feat, const float* __restrict__ a,
                        float* __restrict__ rst, int E) {
  int i = blockIdx.x * blockDim.x + threadIdx.x;
  if (i >= E * NCLS) return;
  int e = i >> 4, c = i & 15;
  atomicAdd(&rst[(size_t)dst[e] * NCLS + c], a[e] * feat[(size_t)src[e] * NCLS + c]);
}

__global__ void k_act0(const float* __restrict__ rst, float* __restrict__ h, int n) {
  int i = blockIdx.x * blockDim.x + threadIdx.x;
  if (i < n) h[i] = eluf(rst[i]);
}
__global__ void k_act1(const float* __restrict__ rst, float* __restrict__ h,
                       float* __restrict__ embOut, int n) {
  int i = blockIdx.x * blockDim.x + threadIdx.x;
  if (i < n) { float v = eluf(rst[i] + h[i]); h[i] = v; embOut[i] = v; }
}
__global__ void k_addres(float* __restrict__ logits, const float* __restrict__ res, int n) {
  int i = blockIdx.x * blockDim.x + threadIdx.x;
  if (i < n) logits[i] += res[i];
}
__global__ void k_norm(const float* __restrict__ logits, float* __restrict__ out, int Nn) {
  int n = blockIdx.x * blockDim.x + threadIdx.x;
  if (n >= Nn) return;
  const float* p = logits + (size_t)n * NCLS;
  float ss = 0.f;
  for (int c = 0; c < NCLS; ++c) ss += p[c] * p[c];
  float inv = 1.f / fmaxf(sqrtf(ss), 1e-12f);
  for (int c = 0; c < NCLS; ++c) out[(size_t)n * NCLS + c] = p[c] * inv;
}

// ---------------------------------------------------------------------------
static inline unsigned gdiv(long long n, int b) { return (unsigned)((n + b - 1) / b); }

extern "C" void kernel_launch(void* const* d_in, const int* in_sizes, int n_in,
                              void* d_out, int out_size, void* d_ws, size_t ws_size,
                              hipStream_t stream) {
  (void)in_sizes; (void)n_in; (void)out_size; (void)ws_size;
  const float* feat0 = (const float*)d_in[0];
  const float* feat1 = (const float*)d_in[1];
  const float* feat2 = (const float*)d_in[2];
  const float* fc0_w = (const float*)d_in[3];
  const float* fc0_b = (const float*)d_in[4];
  const float* fc1_w = (const float*)d_in[5];
  const float* fc1_b = (const float*)d_in[6];
  const float* fc2_w = (const float*)d_in[7];
  const float* fc2_b = (const float*)d_in[8];
  const float* W0    = (const float*)d_in[9];
  const float* eemb0 = (const float*)d_in[10];
  const float* We0   = (const float*)d_in[11];
  const float* al0   = (const float*)d_in[12];
  const float* ar0   = (const float*)d_in[13];
  const float* ae0   = (const float*)d_in[14];
  const float* W1    = (const float*)d_in[15];
  const float* eemb1 = (const float*)d_in[16];
  const float* We1   = (const float*)d_in[17];
  const float* al1   = (const float*)d_in[18];
  const float* ar1   = (const float*)d_in[19];
  const float* ae1   = (const float*)d_in[20];
  const float* Wf    = (const float*)d_in[21];
  const float* eembf = (const float*)d_in[22];
  const float* Wef   = (const float*)d_in[23];
  const float* alf   = (const float*)d_in[24];
  const float* arf   = (const float*)d_in[25];
  const float* aef   = (const float*)d_in[26];
  const float* resfw = (const float*)d_in[27];
  const int* srcI    = (const int*)d_in[28];
  const int* dstI    = (const int*)d_in[29];
  const int* etI     = (const int*)d_in[30];

  float* outLogits = (float*)d_out;
  float* outEmb = outLogits + (size_t)N_NODES * NCLS;

  // ---- workspace carve-out (256B aligned) ----
  char* ws = (char*)d_ws;
  size_t off = 0;
  auto carve = [&](size_t bytes) -> void* {
    void* p = ws + off;
    off = (off + bytes + 255) & ~(size_t)255;
    return p;
  };
  float* bufH    = (float*)carve((size_t)N_NODES * HF * 4);   // node features (layer input)
  float* bufFeat = (float*)carve((size_t)N_NODES * HF * 4);   // projected feat of layer
  float* bufRst  = (float*)carve((size_t)N_NODES * HF * 4);   // aggregation target
  unsigned short* B16 = (unsigned short*)carve((size_t)HF * HF * 2);
  float* el      = (float*)carve((size_t)N_NODES * HEADS * 4);
  float* er      = (float*)carve((size_t)N_NODES * HEADS * 4);
  unsigned* mkey = (unsigned*)carve((size_t)N_NODES * HEADS * 4);
  float* ssum    = (float*)carve((size_t)N_NODES * HEADS * 4);
  float* attnW   = (float*)carve((size_t)N_EDGES * HEADS * 4);
  float* resA    = (float*)carve((size_t)N_EDGES * HEADS * 4);
  float* eeBuf   = (float*)carve(256);
  float* logitsT = (float*)carve((size_t)N_NODES * NCLS * 4);
  // bf16 A copy aliases bufRst (lifetimes disjoint: A16 consumed by GEMM before rst is zeroed)
  unsigned short* A16 = (unsigned short*)bufRst;
  // final-layer residual GEMM output lives past featF inside bufFeat
  float* resOut = bufFeat + (size_t)N_NODES * NCLS;

  hipStream_t s = stream;
  const int TB = 256;

  // ================= input projections -> bufH [N,64] =================
  struct FC { const float* f; const float* w; const float* b; int M; int K; int rowOff; };
  FC fcs[3] = {{feat0, fc0_w, fc0_b, 16000, 256, 0},
               {feat1, fc1_w, fc1_b, 16000, 128, 16000},
               {feat2, fc2_w, fc2_b, 8000, 64, 32000}};
  for (int i = 0; i < 3; ++i) {
    long long nEl = (long long)fcs[i].M * fcs[i].K;
    k_f32_to_bf16<<<gdiv(nEl, TB), TB, 0, s>>>(fcs[i].f, A16, (int)nEl);
    k_transpose_bf16<<<gdiv((long long)HID * fcs[i].K, TB), TB, 0, s>>>(fcs[i].w, B16, fcs[i].K, HID);
    dim3 g(gdiv(fcs[i].M, BM), gdiv(HID, BN));
    k_gemm_bf16<<<g, TB, 0, s>>>(A16, B16, bufH + (size_t)fcs[i].rowOff * HID, fcs[i].b,
                                 fcs[i].M, HID, fcs[i].K);
  }

  // ================= GAT layer 0 (Din=64, act=ELU, save attn) =================
  k_ee<<<N_ETYPES * HEADS, EDGE_DIM, 0, s>>>(eemb0, We0, ae0, eeBuf, EDGE_DIM, HEADS);
  k_f32_to_bf16<<<gdiv((long long)N_NODES * HID, TB), TB, 0, s>>>(bufH, A16, N_NODES * HID);
  k_transpose_bf16<<<gdiv((long long)HF * HID, TB), TB, 0, s>>>(W0, B16, HID, HF);
  { dim3 g(gdiv(N_NODES, BM), gdiv(HF, BN));
    k_gemm_bf16<<<g, TB, 0, s>>>(A16, B16, bufFeat, nullptr, N_NODES, HF, HID); }
  k_elr<<<gdiv((long long)N_NODES * HEADS, TB), TB, 0, s>>>(bufFeat, al0, ar0, el, er, N_NODES, HEADS, HID);
  k_init_ms<<<gdiv((long long)N_NODES * HEADS, TB), TB, 0, s>>>(mkey, ssum, N_NODES * HEADS);
  k_att1<<<gdiv((long long)N_EDGES * HEADS, TB), TB, 0, s>>>(srcI, dstI, etI, el, er, eeBuf, attnW, mkey, N_EDGES, HEADS);
  k_att2<<<gdiv((long long)N_EDGES * HEADS, TB), TB, 0, s>>>(dstI, attnW, mkey, ssum, N_EDGES, HEADS);
  k_att3<<<gdiv((long long)N_EDGES * HEADS, TB), TB, 0, s>>>(dstI, attnW, ssum, nullptr, resA, N_EDGES, HEADS);
  k_zero<<<gdiv((long long)N_NODES * HF, TB), TB, 0, s>>>(bufRst, N_NODES * HF);
  k_agg512<<<gdiv((long long)N_EDGES * 32, TB), TB, 0, s>>>(srcI, dstI, bufFeat, attnW, bufRst, N_EDGES);
  k_act0<<<gdiv((long long)N_NODES * HF, TB), TB, 0, s>>>(bufRst, bufH, N_NODES * HF);

  // ================= GAT layer 1 (Din=512, identity residual, attn blend, ELU) =================
  k_ee<<<N_ETYPES * HEADS, EDGE_DIM, 0, s>>>(eemb1, We1, ae1, eeBuf, EDGE_DIM, HEADS);
  k_f32_to_bf16<<<gdiv((long long)N_NODES * HF, TB), TB, 0, s>>>(bufH, A16, N_NODES * HF);
  k_transpose_bf16<<<gdiv((long long)HF * HF, TB), TB, 0, s>>>(W1, B16, HF, HF);
  { dim3 g(gdiv(N_NODES, BM), gdiv(HF, BN));
    k_gemm_bf16<<<g, TB, 0, s>>>(A16, B16, bufFeat, nullptr, N_NODES, HF, HF); }
  k_elr<<<gdiv((long long)N_NODES * HEADS, TB), TB, 0, s>>>(bufFeat, al1, ar1, el, er, N_NODES, HEADS, HID);
  k_init_ms<<<gdiv((long long)N_NODES * HEADS, TB), TB, 0, s>>>(mkey, ssum, N_NODES * HEADS);
  k_att1<<<gdiv((long long)N_EDGES * HEADS, TB), TB, 0, s>>>(srcI, dstI, etI, el, er, eeBuf, attnW, mkey, N_EDGES, HEADS);
  k_att2<<<gdiv((long long)N_EDGES * HEADS, TB), TB, 0, s>>>(dstI, attnW, mkey, ssum, N_EDGES, HEADS);
  k_att3<<<gdiv((long long)N_EDGES * HEADS, TB), TB, 0, s>>>(dstI, attnW, ssum, resA, nullptr, N_EDGES, HEADS);
  k_zero<<<gdiv((long long)N_NODES * HF, TB), TB, 0, s>>>(bufRst, N_NODES * HF);
  k_agg512<<<gdiv((long long)N_EDGES * 32, TB), TB, 0, s>>>(srcI, dstI, bufFeat, attnW, bufRst, N_EDGES);
  k_act1<<<gdiv((long long)N_NODES * HF, TB), TB, 0, s>>>(bufRst, bufH, outEmb, N_NODES * HF);

  // ================= final layer (H=1, F=NCLS, residual = emb @ resf_w) =================
  k_ee<<<N_ETYPES * 1, EDGE_DIM, 0, s>>>(eembf, Wef, aef, eeBuf, EDGE_DIM, 1);
  k_f32_to_bf16<<<gdiv((long long)N_NODES * HF, TB), TB, 0, s>>>(bufH, A16, N_NODES * HF);
  k_transpose_bf16<<<gdiv((long long)NCLS * HF, TB), TB, 0, s>>>(Wf, B16, HF, NCLS);
  { dim3 g(gdiv(N_NODES, BM), gdiv(NCLS, BN));
    k_gemm_bf16<<<g, TB, 0, s>>>(A16, B16, bufFeat, nullptr, N_NODES, NCLS, HF); }   // featF
  k_transpose_bf16<<<gdiv((long long)NCLS * HF, TB), TB, 0, s>>>(resfw, B16, HF, NCLS);
  { dim3 g(gdiv(N_NODES, BM), gdiv(NCLS, BN));
    k_gemm_bf16<<<g, TB, 0, s>>>(A16, B16, resOut, nullptr, N_NODES, NCLS, HF); }    // emb@resf_w
  k_elr<<<gdiv((long long)N_NODES, TB), TB, 0, s>>>(bufFeat, alf, arf, el, er, N_NODES, 1, NCLS);
  k_init_ms<<<gdiv((long long)N_NODES, TB), TB, 0, s>>>(mkey, ssum, N_NODES);
  k_att1<<<gdiv((long long)N_EDGES, TB), TB, 0, s>>>(srcI, dstI, etI, el, er, eeBuf, attnW, mkey, N_EDGES, 1);
  k_att2<<<gdiv((long long)N_EDGES, TB), TB, 0, s>>>(dstI, attnW, mkey, ssum, N_EDGES, 1);
  k_att3<<<gdiv((long long)N_EDGES, TB), TB, 0, s>>>(dstI, attnW, ssum, nullptr, nullptr, N_EDGES, 1);
  k_zero<<<gdiv((long long)N_NODES * NCLS, TB), TB, 0, s>>>(logitsT, N_NODES * NCLS);
  k_agg16<<<gdiv((long long)N_EDGES * NCLS, TB), TB, 0, s>>>(srcI, dstI, bufFeat, attnW, logitsT, N_EDGES);
  k_addres<<<gdiv((long long)N_NODES * NCLS, TB), TB, 0, s>>>(logitsT, resOut, N_NODES * NCLS);
  k_norm<<<gdiv((long long)N_NODES, TB), TB, 0, s>>>(logitsT, outLogits, N_NODES);
}
